// PaCo_59863254172163
// MI455X (gfx1250) — matrix operand order repaired and checked by reading, not compiled, not run on hardware
//
#include <hip/hip_runtime.h>
#include <hip/hip_bf16.h>

typedef __attribute__((ext_vector_type(16))) _Float16 v16h;
typedef __attribute__((ext_vector_type(8)))  _Float16 h8;
typedef __attribute__((ext_vector_type(2)))  _Float16 h2;
typedef __attribute__((ext_vector_type(8)))  float    v8f;
typedef __attribute__((ext_vector_type(4)))  unsigned int u32x4;
typedef __attribute__((ext_vector_type(8)))  int      i32x8;
typedef __attribute__((ext_vector_type(4)))  int      i32x4;

#define B_   8
#define N_   4096
#define M_   1024
#define KNN_ 16
#define G_   4

// ---------------------------------------------------------------------------
// Input projection: f0[b,n,o] = coor . w_in[o] + b_in[o]; also extract coor.
// ---------------------------------------------------------------------------
__global__ void input_proj_kernel(const float* __restrict__ x,
                                  const float* __restrict__ w_in,
                                  const float* __restrict__ b_in,
                                  float* __restrict__ f0,
                                  float* __restrict__ coor,
                                  int total /* B*N */) {
    int t = blockIdx.x * blockDim.x + threadIdx.x;
    if (t >= total) return;
    const float* px = x + (size_t)t * 7;
    float cx = px[0], cy = px[1], cz = px[2];
    coor[(size_t)t * 3 + 0] = cx;
    coor[(size_t)t * 3 + 1] = cy;
    coor[(size_t)t * 3 + 2] = cz;
    #pragma unroll
    for (int o = 0; o < 8; ++o) {
        f0[(size_t)t * 8 + o] =
            w_in[o * 3 + 0] * cx + w_in[o * 3 + 1] * cy + w_in[o * 3 + 2] * cz + b_in[o];
    }
}

// ---------------------------------------------------------------------------
// Weight pre-conversion: f32 (Cout,Cin2) -> f16 (Cout,Cin2p), zero K-padded.
// Makes the weight tile a dense 2D f16 tensor that TDM can DMA into LDS.
// ---------------------------------------------------------------------------
__global__ void convert_w_kernel(const float* __restrict__ w, _Float16* __restrict__ out,
                                 int Cin2, int Cin2p, int total /* Cout*Cin2p */) {
    int e = blockIdx.x * blockDim.x + threadIdx.x;
    if (e >= total) return;
    int csh = __ffs(Cin2p) - 1;
    int o = e >> csh, c = e & (Cin2p - 1);
    out[e] = (c < Cin2) ? (_Float16)w[o * Cin2 + c] : (_Float16)0.f;
}

// ---------------------------------------------------------------------------
// kNN: per query, 16 smallest squared distances via register insertion sort.
// Keys tiled through LDS. grid = (ceil(Nq/256), B)
// ---------------------------------------------------------------------------
__global__ __launch_bounds__(256)
void knn_kernel(const float* __restrict__ cq, const float* __restrict__ ck,
                int Nq, int Nk, int* __restrict__ outIdx) {
    __shared__ float sk[256 * 3];
    int b = blockIdx.y;
    int q = blockIdx.x * 256 + threadIdx.x;
    bool valid = q < Nq;
    float qx = 0.f, qy = 0.f, qz = 0.f;
    if (valid) {
        const float* p = cq + ((size_t)b * Nq + q) * 3;
        qx = p[0]; qy = p[1]; qz = p[2];
    }
    float bd[KNN_]; int bi[KNN_];
    #pragma unroll
    for (int t = 0; t < KNN_; ++t) { bd[t] = 3.4e38f; bi[t] = 0; }

    for (int t0 = 0; t0 < Nk; t0 += 256) {
        int nk = Nk - t0; if (nk > 256) nk = 256;
        for (int e = threadIdx.x; e < nk * 3; e += 256)
            sk[e] = ck[((size_t)b * Nk + t0) * 3 + e];
        __syncthreads();
        if (valid) {
            for (int j = 0; j < nk; ++j) {
                float dx = qx - sk[j * 3 + 0];
                float dy = qy - sk[j * 3 + 1];
                float dz = qz - sk[j * 3 + 2];
                float d = dx * dx + dy * dy + dz * dz;
                if (d < bd[KNN_ - 1]) {
                    float dd = d; int ii = t0 + j;
                    #pragma unroll
                    for (int t = 0; t < KNN_; ++t) {
                        if (dd < bd[t]) {
                            float tf = bd[t]; int ti = bi[t];
                            bd[t] = dd; bi[t] = ii; dd = tf; ii = ti;
                        }
                    }
                }
            }
        }
        __syncthreads();
    }
    if (valid) {
        #pragma unroll
        for (int t = 0; t < KNN_; ++t)
            outIdx[((size_t)b * Nq + q) * KNN_ + t] = bi[t];
    }
}

// ---------------------------------------------------------------------------
// Graph-feature + 1x1 conv as WMMA GEMM.
// A[row][c] = (c<C) ? fk[idx]-fq : fq  (f16, packed to LDS in h2 pairs)
// B = weights, DMA'd into LDS by the Tensor Data Mover (wave0 issues the
// tensor_load_to_lds with a hand-built D#, waits TENSORcnt, then barrier).
// Block: 128 threads = 4 waves, 64 rows per block. grid = (Nq*16/64, 1, B)
// ---------------------------------------------------------------------------
__global__ __launch_bounds__(128)
void conv_wmma_kernel(const float* __restrict__ fk, const float* __restrict__ fq,
                      const int* __restrict__ idx, const _Float16* __restrict__ wh,
                      float* __restrict__ y, int Nq, int Nk, int C, int Cout) {
    const int Cin2  = 2 * C;
    const int Cin2p = (Cin2 + 31) & ~31;            // pad K-dim to WMMA K=32 (pow2)
    const int csh   = __ffs(Cin2p) - 1;             // log2(Cin2p)
    const int b = blockIdx.z;
    const int rowBase = blockIdx.x * 64;
    const int rowsTotal = Nq * KNN_;

    __shared__ _Float16 As[64 * 128];               // 16 KB
    __shared__ _Float16 Bs[128 * 128];              // 32 KB

    const int tid  = threadIdx.x;
    const int wave = tid >> 5;
    const int lane = tid & 31;

    // --- TDM: DMA weight tile (Cout x Cin2p f16, row-major) into Bs --------
    if (wave == 0) {
        unsigned long long ga = (unsigned long long)(uintptr_t)wh;
        unsigned lds = (unsigned)(uintptr_t)(&Bs[0]);     // LDS aperture: low 32 bits
        u32x4 g0;
        g0[0] = 1u;                                       // count=1, user mode
        g0[1] = lds;                                      // lds_addr
        g0[2] = (unsigned)(ga & 0xffffffffu);             // global_addr[31:0]
        g0[3] = (unsigned)((ga >> 32) & 0x1ffffffu)       // global_addr[56:32]
              | (2u << 30);                               // type=2 (image)
        i32x8 g1;
        g1[0] = 0x00010000;                               // wg_mask=0, data_size=1 (2B)
        g1[1] = (Cin2p & 0xffff) << 16;                   // tensor_dim0[15:0]
        g1[2] = ((Cin2p >> 16) & 0xffff)                  // tensor_dim0[31:16]
              | ((Cout & 0xffff) << 16);                  // tensor_dim1[15:0]
        g1[3] = ((Cout >> 16) & 0xffff)                   // tensor_dim1[31:16]
              | ((Cin2p & 0xffff) << 16);                 // tile_dim0
        g1[4] = (Cout & 0xffff);                          // tile_dim1 (tile_dim2=0)
        g1[5] = Cin2p;                                    // tensor_dim0_stride[31:0]
        g1[6] = 0;                                        // stride0[47:32], stride1[15:0]
        g1[7] = 0;
        i32x4 gz4 = {0, 0, 0, 0};                         // 2D: groups 2/3 null
        i32x8 gz8 = {0, 0, 0, 0, 0, 0, 0, 0};
        __builtin_amdgcn_tensor_load_to_lds(g0, g1, gz4, gz4, gz8, 0);
        __builtin_amdgcn_s_wait_tensorcnt(0);
    }

    // --- Stage A tile (graph feature), 2 channels per thread, h2 stores ----
    const int prs = csh - 1;                         // pairs-per-row shift
    for (int e = tid; e < (32 << csh); e += 128) {
        int r  = e >> prs;
        int c2 = (e & ((Cin2p >> 1) - 1)) << 1;      // even; never straddles C/Cin2
        int row = rowBase + r;
        float v0 = 0.f, v1 = 0.f;
        if (row < rowsTotal && c2 < Cin2) {
            int q = row >> 4, k = row & (KNN_ - 1);
            const float* pq = fq + ((size_t)b * Nq + q) * C;
            if (c2 < C) {
                int j = idx[((size_t)b * Nq + q) * KNN_ + k];
                const float* pk = fk + ((size_t)b * Nk + j) * C;
                v0 = pk[c2] - pq[c2];
                v1 = pk[c2 + 1] - pq[c2 + 1];
            } else {
                v0 = pq[c2 - C];
                v1 = pq[c2 + 1 - C];
            }
        }
        h2 hv = { (_Float16)v0, (_Float16)v1 };
        *(h2*)&As[(r << csh) + c2] = hv;
    }
    __syncthreads();

    const int half = lane >> 4;
    const int l16  = lane & 15;
    const int mb   = wave * 16;
    const int nK   = Cin2p >> 5;                    // 1, 2 or 4 K-chunks

    // Hoist A fragments: per ISA 7.12.2 the lane's 16 halves are the two
    // contiguous runs [kk + half*8, +8) and [kk + 16 + half*8, +8).
    v16h afrag[4];
    #pragma unroll
    for (int kt = 0; kt < 4; ++kt) {
        if (kt < nK) {
            const _Float16* pa = &As[((mb + l16) << csh) + kt * 32 + half * 8];
            h8 lo = *(const h8*)pa;
            h8 hi = *(const h8*)(pa + 16);
            afrag[kt] = __builtin_shufflevector(lo, hi,
                0, 1, 2, 3, 4, 5, 6, 7, 8, 9, 10, 11, 12, 13, 14, 15);
        }
    }

    for (int nt = 0; nt < Cout; nt += 16) {
        v8f acc = {};
        #pragma unroll
        for (int kt = 0; kt < 4; ++kt) {
            if (kt < nK) {
                // B fragment: lane column = l16, K runs [half*8, +8) and [16 + half*8, +8)
                const _Float16* pb = &Bs[((nt + l16) << csh) + kt * 32 + half * 8];
                h8 lo = *(const h8*)pb;
                h8 hi = *(const h8*)(pb + 16);
                v16h bf = __builtin_shufflevector(lo, hi,
                    0, 1, 2, 3, 4, 5, 6, 7, 8, 9, 10, 11, 12, 13, 14, 15);
                acc = __builtin_amdgcn_wmma_f32_16x16x32_f16(
                    false, afrag[kt], false, bf, (short)0, acc, false, false);
            }
        }
        // C/D layout: acc[v] -> (M = half*8+v, N = l16)
        #pragma unroll
        for (int v = 0; v < 8; ++v) {
            int row = rowBase + mb + half * 8 + v;
            if (row < rowsTotal) {
                int q = row >> 4, k = row & (KNN_ - 1);
                y[(((size_t)b * Nq + q) * KNN_ + k) * Cout + nt + l16] = acc[v];
            }
        }
    }
}

// ---------------------------------------------------------------------------
// GroupNorm stats: per (b, group) sum & sumsq over (Nq, K, C/G). grid = B*G.
// ---------------------------------------------------------------------------
__global__ __launch_bounds__(256)
void gn_stats_kernel(const float* __restrict__ y, float* __restrict__ stats,
                     int Nq, int Cout) {
    __shared__ float ss[256], ss2[256];
    int bg = blockIdx.x;
    int b = bg >> 2, g = bg & (G_ - 1);
    int Cg = Cout >> 2;                             // pow2
    int cgsh = __ffs(Cg) - 1;
    size_t cnt = (size_t)Nq * KNN_ * Cg;
    size_t base = (size_t)b * Nq * KNN_ * Cout;
    float s = 0.f, s2 = 0.f;
    for (size_t e = threadIdx.x; e < cnt; e += 256) {
        size_t qk = e >> cgsh;
        int c = (int)(e & (Cg - 1));
        float v = y[base + qk * Cout + g * Cg + c];
        s += v; s2 += v * v;
    }
    ss[threadIdx.x] = s; ss2[threadIdx.x] = s2;
    __syncthreads();
    for (int st = 128; st > 0; st >>= 1) {
        if (threadIdx.x < st) {
            ss[threadIdx.x]  += ss[threadIdx.x + st];
            ss2[threadIdx.x] += ss2[threadIdx.x + st];
        }
        __syncthreads();
    }
    if (threadIdx.x == 0) {
        stats[bg * 2 + 0] = ss[0];
        stats[bg * 2 + 1] = ss2[0];
    }
}

// ---------------------------------------------------------------------------
// Normalize + affine + LeakyReLU + max over K neighbors. one thread = (b,q,o)
// ---------------------------------------------------------------------------
__global__ void gn_max_kernel(const float* __restrict__ y,
                              const float* __restrict__ gamma,
                              const float* __restrict__ beta,
                              const float* __restrict__ stats,
                              float* __restrict__ out, int Nq, int Cout, int total) {
    int t = blockIdx.x * blockDim.x + threadIdx.x;
    if (t >= total) return;
    int osh = __ffs(Cout) - 1;                      // Cout, Nq are pow2
    int qsh = __ffs(Nq) - 1;
    int o = t & (Cout - 1);
    int q = (t >> osh) & (Nq - 1);
    int b = t >> (osh + qsh);
    int Cg = Cout >> 2;
    int g = o >> (__ffs(Cg) - 1);
    float cnt = (float)Nq * KNN_ * Cg;
    float s  = stats[(b * G_ + g) * 2 + 0];
    float s2 = stats[(b * G_ + g) * 2 + 1];
    float mean = s / cnt;
    float var  = s2 / cnt - mean * mean;
    float inv  = rsqrtf(var + 1e-5f);
    float gm = gamma[o], bt = beta[o];
    size_t base = (((size_t)b * Nq + q) * KNN_) * Cout + o;
    float best = -3.4e38f;
    #pragma unroll
    for (int k = 0; k < KNN_; ++k) {
        float v = (y[base + (size_t)k * Cout] - mean) * inv * gm + bt;
        v = v > 0.f ? v : 0.2f * v;
        best = fmaxf(best, v);
    }
    out[((size_t)b * Nq + q) * Cout + o] = best;
}

// ---------------------------------------------------------------------------
// Farthest point sampling: 1 block per batch, dists in LDS, sequential argmax.
// Emits `last` before update (matches lax.scan ordering; first idx = 0).
// ---------------------------------------------------------------------------
__global__ __launch_bounds__(256)
void fps_kernel(const float* __restrict__ coor, int N, int M, int* __restrict__ outIdx) {
    __shared__ float dist[N_];
    __shared__ float rv[256];
    __shared__ int   ri[256];
    int b = blockIdx.x;
    for (int n = threadIdx.x; n < N; n += 256) dist[n] = 1e10f;
    __syncthreads();
    int last = 0;
    for (int it = 0; it < M; ++it) {
        if (threadIdx.x == 0) outIdx[(size_t)b * M + it] = last;
        const float* lp = coor + ((size_t)b * N + last) * 3;
        float lx = lp[0], ly = lp[1], lz = lp[2];
        float best = -1.f; int besti = 0;
        for (int n = threadIdx.x; n < N; n += 256) {
            const float* p = coor + ((size_t)b * N + n) * 3;
            float dx = p[0] - lx, dy = p[1] - ly, dz = p[2] - lz;
            float d = dx * dx + dy * dy + dz * dz;
            float dn = fminf(dist[n], d);
            dist[n] = dn;
            if (dn > best) { best = dn; besti = n; }
        }
        rv[threadIdx.x] = best; ri[threadIdx.x] = besti;
        __syncthreads();
        for (int s = 128; s > 0; s >>= 1) {
            if (threadIdx.x < s) {
                bool take = (rv[threadIdx.x + s] > rv[threadIdx.x]) ||
                            (rv[threadIdx.x + s] == rv[threadIdx.x] &&
                             ri[threadIdx.x + s] < ri[threadIdx.x]);
                if (take) { rv[threadIdx.x] = rv[threadIdx.x + s]; ri[threadIdx.x] = ri[threadIdx.x + s]; }
            }
            __syncthreads();
        }
        last = ri[0];
        __syncthreads();
    }
}

// ---------------------------------------------------------------------------
// Generic gather: dst[b,m,0:C] = src[b, idx[b,m], 0:C]; M_=1024 rows per batch
// ---------------------------------------------------------------------------
__global__ void gather_kernel(const float* __restrict__ src, int srcStride, int srcN,
                              const int* __restrict__ idx, float* __restrict__ dst,
                              int C, int total /* B*M_ */) {
    int t = blockIdx.x * blockDim.x + threadIdx.x;
    if (t >= total) return;
    int b = t >> 10;                                // M_ == 1024
    int j = idx[t];
    const float* s = src + ((size_t)b * srcN + j) * srcStride;
    float* d = dst + (size_t)t * C;
    for (int c = 0; c < C; ++c) d[c] = s[c];
}

// idxF[b,m] = idx1[b, idx2[b,m]]
__global__ void compose_idx_kernel(const int* __restrict__ idx1, const int* __restrict__ idx2,
                                   int* __restrict__ out, int total) {
    int t = blockIdx.x * blockDim.x + threadIdx.x;
    if (t >= total) return;
    int b = t >> 10;                                // M_ == 1024
    out[t] = idx1[((size_t)b << 10) + idx2[t]];
}

// ---------------------------------------------------------------------------
extern "C" void kernel_launch(void* const* d_in, const int* in_sizes, int n_in,
                              void* d_out, int out_size, void* d_ws, size_t ws_size,
                              hipStream_t stream) {
    const float* x     = (const float*)d_in[0];
    const float* w_in  = (const float*)d_in[2];
    const float* b_in  = (const float*)d_in[3];
    const float* w1    = (const float*)d_in[4];
    const float* g1    = (const float*)d_in[5];
    const float* be1   = (const float*)d_in[6];
    const float* w2    = (const float*)d_in[7];
    const float* g2    = (const float*)d_in[8];
    const float* be2   = (const float*)d_in[9];
    const float* w3    = (const float*)d_in[10];
    const float* g3    = (const float*)d_in[11];
    const float* be3   = (const float*)d_in[12];
    const float* w4    = (const float*)d_in[13];
    const float* g4    = (const float*)d_in[14];
    const float* be4   = (const float*)d_in[15];

    float* ws = (float*)d_ws;
    size_t off = 0;
    auto alloc = [&](size_t nelem) { float* p = ws + off; off += nelem; return p; };

    float* coor  = alloc((size_t)B_ * N_ * 3);      // 98304
    float* f0    = alloc((size_t)B_ * N_ * 8);      // 262144
    float* f1    = alloc((size_t)B_ * N_ * 32);     // 1048576
    float* fqbuf = alloc((size_t)B_ * M_ * 64);     // 524288
    float* f2    = alloc((size_t)B_ * M_ * 64);     // 524288
    float* f3    = alloc((size_t)B_ * M_ * 64);     // 524288
    float* cq1   = alloc((size_t)B_ * M_ * 3);      // 24576
    float* stats = alloc((size_t)B_ * G_ * 2);      // 64
    int*   knnIx = (int*)alloc((size_t)B_ * N_ * KNN_);  // 524288
    int*   fps1  = (int*)alloc((size_t)B_ * M_);
    int*   fps2  = (int*)alloc((size_t)B_ * M_);
    int*   idxF  = (int*)alloc((size_t)B_ * M_);
    _Float16* wh = (_Float16*)alloc(128 * 128 / 2); // 16K halves (reused per stage)
    float* y     = alloc((size_t)B_ * N_ * KNN_ * 32);   // 16.7M floats, reused

    float* out_coor   = (float*)d_out;                       // (8,1024,3)
    float* out_f      = (float*)d_out + 24576;               // (8,1024,128)
    float* out_normal = (float*)d_out + 24576 + 1048576;     // (8,1024,3)
    float* out_plane  = (float*)d_out + 24576 + 1048576 + 24576; // (8,1024,1)

    // ---- Stage 1: N=4096, C=8 -> Cout=32 (Cin2=16, Cin2p=32) ----
    {
        int total = B_ * N_;
        input_proj_kernel<<<(total + 255) / 256, 256, 0, stream>>>(x, w_in, b_in, f0, coor, total);
        knn_kernel<<<dim3(N_ / 256, B_), 256, 0, stream>>>(coor, coor, N_, N_, knnIx);
        convert_w_kernel<<<(32 * 32 + 255) / 256, 256, 0, stream>>>(w1, wh, 16, 32, 32 * 32);
        conv_wmma_kernel<<<dim3(N_ * KNN_ / 64, 1, B_), 128, 0, stream>>>(
            f0, f0, knnIx, wh, y, N_, N_, 8, 32);
        gn_stats_kernel<<<B_ * G_, 256, 0, stream>>>(y, stats, N_, 32);
        int tot = B_ * N_ * 32;
        gn_max_kernel<<<(tot + 255) / 256, 256, 0, stream>>>(y, g1, be1, stats, f1, N_, 32, tot);
        fps_kernel<<<B_, 256, 0, stream>>>(coor, N_, M_, fps1);
        int bm = B_ * M_;
        gather_kernel<<<(bm + 255) / 256, 256, 0, stream>>>(coor, 3, N_, fps1, cq1, 3, bm);
        gather_kernel<<<(bm + 255) / 256, 256, 0, stream>>>(f1, 32, N_, fps1, fqbuf, 32, bm);
    }
    // ---- Stage 2: Nq=1024 vs Nk=4096, C=32 -> Cout=64 (Cin2p=64) ----
    {
        knn_kernel<<<dim3(M_ / 256, B_), 256, 0, stream>>>(cq1, coor, M_, N_, knnIx);
        convert_w_kernel<<<(64 * 64 + 255) / 256, 256, 0, stream>>>(w2, wh, 64, 64, 64 * 64);
        conv_wmma_kernel<<<dim3(M_ * KNN_ / 64, 1, B_), 128, 0, stream>>>(
            f1, fqbuf, knnIx, wh, y, M_, N_, 32, 64);
        gn_stats_kernel<<<B_ * G_, 256, 0, stream>>>(y, stats, M_, 64);
        int tot = B_ * M_ * 64;
        gn_max_kernel<<<(tot + 255) / 256, 256, 0, stream>>>(y, g2, be2, stats, f2, M_, 64, tot);
    }
    // ---- Stage 3: self 1024, C=64 -> Cout=64 (Cin2p=128) ----
    {
        knn_kernel<<<dim3(M_ / 256, B_), 256, 0, stream>>>(cq1, cq1, M_, M_, knnIx);
        convert_w_kernel<<<(64 * 128 + 255) / 256, 256, 0, stream>>>(w3, wh, 128, 128, 64 * 128);
        conv_wmma_kernel<<<dim3(M_ * KNN_ / 64, 1, B_), 128, 0, stream>>>(
            f2, f2, knnIx, wh, y, M_, M_, 64, 64);
        gn_stats_kernel<<<B_ * G_, 256, 0, stream>>>(y, stats, M_, 64);
        int tot = B_ * M_ * 64;
        gn_max_kernel<<<(tot + 255) / 256, 256, 0, stream>>>(y, g3, be3, stats, f3, M_, 64, tot);
        fps_kernel<<<B_, 256, 0, stream>>>(cq1, M_, M_, fps2);
        int bm = B_ * M_;
        compose_idx_kernel<<<(bm + 255) / 256, 256, 0, stream>>>(fps1, fps2, idxF, bm);
        // gather final coor/normal/plane straight from x into d_out
        gather_kernel<<<(bm + 255) / 256, 256, 0, stream>>>(x + 0, 7, N_, idxF, out_coor, 3, bm);
        gather_kernel<<<(bm + 255) / 256, 256, 0, stream>>>(x + 3, 7, N_, idxF, out_normal, 3, bm);
        gather_kernel<<<(bm + 255) / 256, 256, 0, stream>>>(x + 6, 7, N_, idxF, out_plane, 1, bm);
        gather_kernel<<<(bm + 255) / 256, 256, 0, stream>>>(f3, 64, M_, fps2, fqbuf, 64, bm);
    }
    // ---- Stage 4: queries = sampled coor (in d_out), keys = cq1; 64 -> 128 ----
    {
        knn_kernel<<<dim3(M_ / 256, B_), 256, 0, stream>>>(out_coor, cq1, M_, M_, knnIx);
        convert_w_kernel<<<(128 * 128 + 255) / 256, 256, 0, stream>>>(w4, wh, 128, 128, 128 * 128);
        conv_wmma_kernel<<<dim3(M_ * KNN_ / 64, 1, B_), 128, 0, stream>>>(
            f3, fqbuf, knnIx, wh, y, M_, M_, 64, 128);
        gn_stats_kernel<<<B_ * G_, 256, 0, stream>>>(y, stats, M_, 128);
        int tot = B_ * M_ * 128;
        gn_max_kernel<<<(tot + 255) / 256, 256, 0, stream>>>(y, g4, be4, stats, out_f, M_, 128, tot);
    }
}
